// SoftNearestNeighbourLoss_12627203850286
// MI455X (gfx1250) — matrix operand order repaired and checked by reading, not compiled, not run on hardware
//
#include <hip/hip_runtime.h>
#include <hip/hip_bf16.h>
#include <math.h>

typedef __attribute__((ext_vector_type(16))) _Float16 v16h;
typedef __attribute__((ext_vector_type(8)))  float    v8f;
typedef __attribute__((ext_vector_type(4)))  unsigned int u32x4;  // POD 16B (union-safe)
typedef __attribute__((ext_vector_type(4)))  float        f32x4;

#define BATCH 8192
#define DIM   2048
#define TEMP_INV 2.0f      // e = exp(-sim)/0.5 = 2*exp(-sim)
#define EPS   1e-10f
#define TILE  128          // 128x128 output tile per block
#define KSTEP 64           // two f16-WMMA K-slabs per staged panel
#define NSTAGE (DIM / KSTEP)
#define STG_U4 1024        // u32x4 per stage per panel (128*64/8)
#define BLOCK_THREADS 256  // 8 wave32s: 4 in M x 2 in N, each wave 32x64

// ---------------- zero accumulators ----------------
__global__ void snn_zero(float* __restrict__ p, int n) {
    int i = blockIdx.x * blockDim.x + threadIdx.x;
    if (i < n) p[i] = 0.0f;
}

// ---------------- row L2-normalize fp32 -> f16 ----------------
__global__ __launch_bounds__(256) void snn_normalize(const float* __restrict__ x,
                                                     _Float16* __restrict__ xn) {
    const int row = blockIdx.x;
    const int t   = threadIdx.x;
    const f32x4* xr = (const f32x4*)(x + (size_t)row * DIM);
    f32x4 a = xr[2 * t + 0];
    f32x4 b = xr[2 * t + 1];
    float ss = a.x*a.x + a.y*a.y + a.z*a.z + a.w*a.w
             + b.x*b.x + b.y*b.y + b.z*b.z + b.w*b.w;

    __shared__ float red[256];
    red[t] = ss;
    __syncthreads();
    for (int s = 128; s > 0; s >>= 1) {
        if (t < s) red[t] += red[t + s];
        __syncthreads();
    }
    const float rn = rsqrtf(red[0]);

    union { _Float16 h[8]; u32x4 u; } o;
    o.h[0] = (_Float16)(a.x * rn); o.h[1] = (_Float16)(a.y * rn);
    o.h[2] = (_Float16)(a.z * rn); o.h[3] = (_Float16)(a.w * rn);
    o.h[4] = (_Float16)(b.x * rn); o.h[5] = (_Float16)(b.y * rn);
    o.h[6] = (_Float16)(b.z * rn); o.h[7] = (_Float16)(b.w * rn);
    ((u32x4*)(xn + (size_t)row * DIM))[t] = o.u;
}

// ---------------- fused symmetric GEMM + exp/mask/row-sum ----------------
// 128x128 tile per block, upper-triangular tiles only (jt >= it).
// Double-buffered 128x64 LDS panels -> one barrier per 64-deep K-step.
__global__ __launch_bounds__(BLOCK_THREADS) void snn_sim(const _Float16* __restrict__ xn,
                                                         const int*      __restrict__ y,
                                                         float* __restrict__ gE,
                                                         float* __restrict__ gP) {
    const int it = blockIdx.y, jt = blockIdx.x;
    if (jt < it) return;                    // symmetry: upper triangle only
    const int  ibase = it * TILE, jbase = jt * TILE;
    const bool diag  = (it == jt);
    const int  t     = threadIdx.x;
    const int  lane  = t & 31;
    const int  w     = t >> 5;              // wave32 id, 0..7
    const int  wm    = w >> 1;              // 0..3 : 32-row strip
    const int  wn    = w & 1;               // 0..1 : 64-col strip
    const int  m     = lane & 15;           // 16-lane sub-id
    const int  hi    = lane >> 4;           // half-wave select

    __shared__ _Float16 As[2][TILE * KSTEP];   // 2 x 16 KB
    __shared__ _Float16 Bs[2][TILE * KSTEP];   // 2 x 16 KB
    __shared__ int   sYi[TILE], sYj[TILE];
    __shared__ float sRowE[TILE], sRowP[TILE];
    __shared__ float sColE[TILE], sColP[TILE];

    if (t < TILE) { sYi[t] = y[ibase + t]; sRowE[t] = 0.0f; sRowP[t] = 0.0f; }
    else { int u = t - TILE; sYj[u] = y[jbase + u]; sColE[u] = 0.0f; sColP[u] = 0.0f; }

    v8f acc[2][4] = {};                     // 32(M) x 64(N) per wave

    u32x4* As4 = (u32x4*)As;
    u32x4* Bs4 = (u32x4*)Bs;

    // cooperative staging: thread t covers halves [32t, 32t+32) of a 128x64 panel
    const int lrow = t >> 1;                // 0..127
    const int loff = (t & 1) * 4;           // u32x4 offset within the 8-u32x4 row
    const u32x4* gA = (const u32x4*)(xn + (size_t)(ibase + lrow) * DIM);
    const u32x4* gB = (const u32x4*)(xn + (size_t)(jbase + lrow) * DIM);
    const int sidx = lrow * 8 + loff;       // u32x4 index within a stage

    // fragment base indices (u32x4 units within a stage; row stride = 8 u32x4)
    const int arow0 = (wm * 32 + m) * 8;             // A rows for a=0 (a=1: +16 rows = +128)
    const int brow0 = (wn * 64 + m) * 8;             // B rows for f=0 (f+1: +128)

    // prologue: fill stage 0
    {
        u32x4 a0 = gA[loff], a1 = gA[loff + 1], a2 = gA[loff + 2], a3 = gA[loff + 3];
        u32x4 b0 = gB[loff], b1 = gB[loff + 1], b2 = gB[loff + 2], b3 = gB[loff + 3];
        As4[sidx] = a0; As4[sidx + 1] = a1; As4[sidx + 2] = a2; As4[sidx + 3] = a3;
        Bs4[sidx] = b0; Bs4[sidx + 1] = b1; Bs4[sidx + 2] = b2; Bs4[sidx + 3] = b3;
    }

    union Frag { u32x4 u[2]; v16h h; };

    #pragma unroll 1
    for (int s = 0; s < NSTAGE; ++s) {
        u32x4 a0, a1, a2, a3, b0, b1, b2, b3;
        const bool more = (s + 1 < NSTAGE);
        if (more) {                          // issue next-stage loads before barrier
            const int gidx = (s + 1) * 8 + loff;
            a0 = gA[gidx]; a1 = gA[gidx + 1]; a2 = gA[gidx + 2]; a3 = gA[gidx + 3];
            b0 = gB[gidx]; b1 = gB[gidx + 1]; b2 = gB[gidx + 2]; b3 = gB[gidx + 3];
        }
        __syncthreads();                     // stage s&1 visible; prior reads of (s+1)&1 done
        if (more) {
            const int d = ((s + 1) & 1) * STG_U4 + sidx;
            As4[d] = a0; As4[d + 1] = a1; As4[d + 2] = a2; As4[d + 3] = a3;
            Bs4[d] = b0; Bs4[d + 1] = b1; Bs4[d + 2] = b2; Bs4[d + 3] = b3;
        }

        const int sb = (s & 1) * STG_U4;
        #pragma unroll
        for (int ks = 0; ks < 2; ++ks) {     // two K=32 slabs per staged panel
            const int kb = ks * 4;           // u32x4 offset of slab within a row
            Frag fa0, fa1;
            fa0.u[0] = As4[sb + arow0 + kb + hi];            // K = hi*8 .. +8
            fa0.u[1] = As4[sb + arow0 + kb + 2 + hi];        // K = 16+hi*8 .. +8
            fa1.u[0] = As4[sb + arow0 + 128 + kb + hi];
            fa1.u[1] = As4[sb + arow0 + 128 + kb + 2 + hi];
            #pragma unroll
            for (int f = 0; f < 4; ++f) {
                Frag bf;                     // lane = col N, half hi -> K = hi*8..+8, 16+hi*8..+8
                bf.u[0] = Bs4[sb + brow0 + f * 128 + kb + hi];
                bf.u[1] = Bs4[sb + brow0 + f * 128 + kb + 2 + hi];
                acc[0][f] = __builtin_amdgcn_wmma_f32_16x16x32_f16(
                    false, fa0.h, false, bf.h, (short)0, acc[0][f], false, false);
                acc[1][f] = __builtin_amdgcn_wmma_f32_16x16x32_f16(
                    false, fa1.h, false, bf.h, (short)0, acc[1][f], false, false);
            }
        }
    }

    // epilogue: C layout = (VGPR r, lanes 0-15 -> M=r, lanes 16-31 -> M=r+8; N = lane&15)
    float rE[2][8], rP[2][8];
    #pragma unroll
    for (int a = 0; a < 2; ++a)
        #pragma unroll
        for (int r = 0; r < 8; ++r) { rE[a][r] = 0.0f; rP[a][r] = 0.0f; }
    float cE[4] = {0.f, 0.f, 0.f, 0.f}, cP[4] = {0.f, 0.f, 0.f, 0.f};

    #pragma unroll
    for (int f = 0; f < 4; ++f) {
        const int jloc = wn * 64 + f * 16 + m;
        const int j    = jbase + jloc;
        const int yj   = sYj[jloc];
        #pragma unroll
        for (int a = 0; a < 2; ++a) {
            #pragma unroll
            for (int r = 0; r < 8; ++r) {
                const int   iloc = wm * 32 + a * 16 + hi * 8 + r;
                const int   i    = ibase + iloc;
                const float e    = TEMP_INV * __expf(-acc[a][f][r]);
                const bool  off  = (i != j);
                const float eo   = off ? e : 0.0f;
                const float ep   = (off && (sYi[iloc] == yj)) ? e : 0.0f;
                rE[a][r] += eo; rP[a][r] += ep;
                cE[f] += eo;    cP[f] += ep;
            }
        }
    }

    // row sums: reduce across the 16 lanes of each half-wave (wn=0/1 waves contend)
    #pragma unroll
    for (int a = 0; a < 2; ++a) {
        #pragma unroll
        for (int r = 0; r < 8; ++r) {
            float e = rE[a][r], p = rP[a][r];
            #pragma unroll
            for (int msk = 1; msk < 16; msk <<= 1) {
                e += __shfl_xor(e, msk, 32);
                p += __shfl_xor(p, msk, 32);
            }
            if (m == 0) {
                const int iloc = wm * 32 + a * 16 + hi * 8 + r;
                atomicAdd(&sRowE[iloc], e);
                atomicAdd(&sRowP[iloc], p);
            }
        }
    }
    // column sums (mirror contribution e[j][i] = e[i][j]) for off-diagonal tiles
    if (!diag) {
        #pragma unroll
        for (int f = 0; f < 4; ++f) {
            const float e = cE[f] + __shfl_xor(cE[f], 16, 32);
            const float p = cP[f] + __shfl_xor(cP[f], 16, 32);
            if (hi == 0) {                  // 4 wm-waves hit same slots -> LDS atomics
                atomicAdd(&sColE[wn * 64 + f * 16 + m], e);
                atomicAdd(&sColP[wn * 64 + f * 16 + m], p);
            }
        }
    }
    __syncthreads();

    if (t < TILE) {
        atomicAdd(&gE[ibase + t], sRowE[t]);
        atomicAdd(&gP[ibase + t], sRowP[t]);
    } else if (!diag) {
        const int u = t - TILE;
        atomicAdd(&gE[jbase + u], sColE[u]);
        atomicAdd(&gP[jbase + u], sColP[u]);
    }
}

// ---------------- final scalar reduction ----------------
__global__ __launch_bounds__(256) void snn_finalize(const float* __restrict__ gE,
                                                    const float* __restrict__ gP,
                                                    float* __restrict__ out) {
    const int t = threadIdx.x;
    float s = 0.0f;
    for (int i = t; i < BATCH; i += 256)
        s += logf((gP[i] + EPS) / (gE[i] + EPS));
    __shared__ float red[256];
    red[t] = s;
    __syncthreads();
    for (int k = 128; k > 0; k >>= 1) {
        if (t < k) red[t] += red[t + k];
        __syncthreads();
    }
    if (t == 0) out[0] = -red[0] / (float)BATCH;
}

extern "C" void kernel_launch(void* const* d_in, const int* in_sizes, int n_in,
                              void* d_out, int out_size, void* d_ws, size_t ws_size,
                              hipStream_t stream) {
    const float* x = (const float*)d_in[0];
    const int*   y = (const int*)d_in[1];

    char* ws = (char*)d_ws;
    _Float16* xn = (_Float16*)ws;                              // 32 MB f16 normalized rows
    float* gE = (float*)(ws + (size_t)BATCH * DIM * 2);        // 8192 off-diag sums
    float* gP = gE + BATCH;                                    // 8192 same-label sums

    snn_zero<<<(2 * BATCH + 255) / 256, 256, 0, stream>>>(gE, 2 * BATCH);
    snn_normalize<<<BATCH, 256, 0, stream>>>(x, xn);

    dim3 grid(BATCH / TILE, BATCH / TILE);                     // upper-tri blocks run
    snn_sim<<<grid, BLOCK_THREADS, 0, stream>>>(xn, y, gE, gP);

    snn_finalize<<<1, 256, 0, stream>>>(gE, gP, (float*)d_out);
}
